// MogrifierLSTMClassifier_79156247265963
// MI455X (gfx1250) — compile-verified, matrix operand order checked by reading, not run on hardware
//
#include <hip/hip_runtime.h>
#include <hip/hip_bf16.h>

// ---------------------------------------------------------------------------
// Mogrifier-LSTM classifier for MI455X (gfx1250, wave32, WMMA).
//
// Strategy: single persistent kernel over the T=256 recurrence with software
// grid barriers; all matmuls in bf16 WMMA (v_wmma_f32_16x16x32_bf16) with
// fp32 accumulation; weights converted+transposed to bf16 [N,K] once per
// launch so A and B fragments are two 16B contiguous loads per lane.
// Weights (14 MB bf16) and recurrent state live in the 192 MB L2.
// ---------------------------------------------------------------------------

typedef __bf16 v16bf __attribute__((ext_vector_type(16)));
typedef float  v8f   __attribute__((ext_vector_type(8)));

namespace {
constexpr int B_  = 64;
constexpr int T_  = 256;
constexpr int I_  = 512;
constexpr int H_  = 1024;
constexpr int G4_ = 4096;   // 4*H
constexpr int C_  = 2;

constexpr int NWG         = 32;            // persistent workgroups
constexpr int NTHR        = 256;           // 8 wave32 per WG
constexpr int TOTAL_WAVES = NWG * (NTHR / 32);
}

// ---------------- device helpers ----------------

__device__ __forceinline__ float sigmoidf_(float v) {
    return 1.0f / (1.0f + __expf(-v));
}
__device__ __forceinline__ float tanhf_(float v) {
    float e = __expf(-2.0f * v);
    return (1.0f - e) / (1.0f + e);
}

// K-loop of 16x16x32 bf16 WMMAs. A row-major [M,K], Bt row-major [N,K]
// (i.e. B pre-transposed). Per-lane fragment = two 16B loads: K blocks
// [kb, kb+8) and [kb+16, kb+24), kb = 0 (lanes 0-15) or 8 (lanes 16-31),
// matching the CDNA5 16-bit A-matrix VGPR layout.
__device__ __forceinline__ v8f bf16_kloop(const __bf16* __restrict__ Ar,
                                          const __bf16* __restrict__ Br,
                                          int K, v8f acc) {
    for (int k0 = 0; k0 < K; k0 += 32) {
        union { uint4 u[2]; v16bf v; } fa, fb;
        const uint4* ap = (const uint4*)(Ar + k0);
        const uint4* bp = (const uint4*)(Br + k0);
        fa.u[0] = ap[0]; fa.u[1] = ap[2];
        fb.u[0] = bp[0]; fb.u[1] = bp[2];
        acc = __builtin_amdgcn_wmma_f32_16x16x32_bf16(
            /*neg_a=*/false, fa.v, /*neg_b=*/false, fb.v,
            /*c_mod=*/(short)0, acc, /*reuse_a=*/false, /*reuse_b=*/false);
    }
    return acc;
}

// D[M,N] = A[M,K] * Bt[N,K]^T, wave-tiled 16x16, fp32 out.
__device__ __forceinline__ void mm_bf16(const __bf16* __restrict__ A,
                                        const __bf16* __restrict__ Bt,
                                        float* __restrict__ D,
                                        int M, int N, int K, int wave) {
    const int lane = (int)(threadIdx.x & 31u);
    const int rin  = lane & 15;
    const int kb   = (lane < 16) ? 0 : 8;
    const int ntn    = N >> 4;
    const int ntiles = (M >> 4) * ntn;
    for (int tile = wave; tile < ntiles; tile += TOTAL_WAVES) {
        const int mt = tile / ntn, nt = tile % ntn;
        const __bf16* Ar = A  + (size_t)(mt * 16 + rin) * K + kb;
        const __bf16* Br = Bt + (size_t)(nt * 16 + rin) * K + kb;
        __builtin_prefetch(Bt + (size_t)((nt * 16 + rin + 16) % N) * K, 0, 0);
        v8f acc = {};
        acc = bf16_kloop(Ar, Br, K, acc);
        // D layout: VGPR r -> row mbase + r (+8 for lanes 16-31), col nbase+rin
        float* Dp = D + (size_t)(mt * 16 + ((lane < 16) ? 0 : 8)) * N + nt * 16 + rin;
#pragma unroll
        for (int r = 0; r < 8; ++r) Dp[(size_t)r * N] = acc[r];
    }
}

// gates[B,4H] = xt[B,I]*WxT^T + h[B,H]*WhT^T  (bias folded in later)
__device__ __forceinline__ void mm_gates(const __bf16* __restrict__ xtbf,
                                         const __bf16* __restrict__ WxT,
                                         const __bf16* __restrict__ hbf,
                                         const __bf16* __restrict__ WhT,
                                         float* __restrict__ gates, int wave) {
    const int lane = (int)(threadIdx.x & 31u);
    const int rin  = lane & 15;
    const int kb   = (lane < 16) ? 0 : 8;
    const int ntn    = G4_ >> 4;
    const int ntiles = (B_ >> 4) * ntn;
    for (int tile = wave; tile < ntiles; tile += TOTAL_WAVES) {
        const int mt = tile / ntn, nt = tile % ntn;
        v8f acc = {};
        acc = bf16_kloop(xtbf + (size_t)(mt * 16 + rin) * I_ + kb,
                         WxT  + (size_t)(nt * 16 + rin) * I_ + kb, I_, acc);
        acc = bf16_kloop(hbf  + (size_t)(mt * 16 + rin) * H_ + kb,
                         WhT  + (size_t)(nt * 16 + rin) * H_ + kb, H_, acc);
        float* Dp = gates + (size_t)(mt * 16 + ((lane < 16) ? 0 : 8)) * G4_ + nt * 16 + rin;
#pragma unroll
        for (int r = 0; r < 8; ++r) Dp[(size_t)r * G4_] = acc[r];
    }
}

// Software grid barrier (persistent kernel). bar[0]=arrive count, bar[1]=gen.
__device__ __forceinline__ void gbar(unsigned* bar) {
    __syncthreads();
    if (threadIdx.x == 0) {
        __threadfence();
        volatile unsigned* vb = (volatile unsigned*)bar;
        unsigned gen  = vb[1];
        unsigned prev = atomicAdd(&bar[0], 1u);
        if (prev == (unsigned)(gridDim.x - 1)) {
            vb[0] = 0;
            __threadfence();
            atomicAdd(&bar[1], 1u);
        } else {
            while (vb[1] == gen) { __builtin_amdgcn_s_sleep(1); }
        }
        __threadfence();
    }
    __syncthreads();
}

// ---------------- prep: bf16 convert + transpose weights, zero state ----------------

__global__ void mog_prep(const float* __restrict__ Wx, const float* __restrict__ Wh,
                         const float* __restrict__ Q,  const float* __restrict__ R,
                         __bf16* __restrict__ WxT, __bf16* __restrict__ WhT,
                         __bf16* __restrict__ QT,  __bf16* __restrict__ RT,
                         float* __restrict__ h, float* __restrict__ c,
                         __bf16* __restrict__ hbf, unsigned* __restrict__ bar) {
    const long tid = (long)blockIdx.x * blockDim.x + threadIdx.x;
    const long nth = (long)gridDim.x * blockDim.x;
    if (tid == 0) { bar[0] = 0u; bar[1] = 0u; }
    // WxT[n,k] = Wx[k,n]  (n<4H, k<I)
    for (long e = tid; e < (long)G4_ * I_; e += nth) {
        int n = (int)(e / I_), k = (int)(e % I_);
        WxT[e] = (__bf16)Wx[(long)k * G4_ + n];
    }
    // WhT[n,k] = Wh[k,n]  (n<4H, k<H)
    for (long e = tid; e < (long)G4_ * H_; e += nth) {
        int n = (int)(e / H_), k = (int)(e % H_);
        WhT[e] = (__bf16)Wh[(long)k * G4_ + n];
    }
    // QT[n,k] = Q[k,n]    (n<I, k<H)   for h@Q
    for (long e = tid; e < (long)I_ * H_; e += nth) {
        int n = (int)(e / H_), k = (int)(e % H_);
        QT[e] = (__bf16)Q[(long)k * I_ + n];
    }
    // RT[n,k] = R[k,n]    (n<H, k<I)   for xt@R
    for (long e = tid; e < (long)H_ * I_; e += nth) {
        int n = (int)(e / I_), k = (int)(e % I_);
        RT[e] = (__bf16)R[(long)k * H_ + n];
    }
    for (long e = tid; e < (long)B_ * H_; e += nth) {
        h[e] = 0.0f; c[e] = 0.0f; hbf[e] = (__bf16)0.0f;
    }
}

// ---------------- persistent recurrence kernel ----------------

__global__ __launch_bounds__(NTHR, 1)
void mog_lstm_persistent(const float* __restrict__ x, const float* __restrict__ bias,
                         const float* __restrict__ fcW, const float* __restrict__ fcb,
                         const __bf16* __restrict__ WxT, const __bf16* __restrict__ WhT,
                         const __bf16* __restrict__ QT,  const __bf16* __restrict__ RT,
                         __bf16* __restrict__ hbf, __bf16* __restrict__ xtbf,
                         float* __restrict__ h, float* __restrict__ c,
                         float* __restrict__ xt, float* __restrict__ tmp,
                         float* __restrict__ gates, unsigned* __restrict__ bar,
                         float* __restrict__ out) {
    const int tid  = (int)(blockIdx.x * blockDim.x + threadIdx.x);
    const int nth  = (int)(gridDim.x * blockDim.x);
    const int wave = tid >> 5;
    const int lane = (int)(threadIdx.x & 31u);

    for (int t = 0; t < T_; ++t) {
        // ---- 5 mogrification rounds ----
        for (int r = 0; r < 5; ++r) {
            if ((r & 1) == 0) {
                // tmp[B,I] = h @ Q  (K=H)
                mm_bf16(hbf, QT, tmp, B_, I_, H_, wave);
                gbar(bar);
                const float* src    = (r == 0) ? (x + (size_t)t * I_) : xt;
                const int    stride = (r == 0) ? (T_ * I_) : I_;
                for (int e = tid; e < B_ * I_; e += nth) {
                    int b = e / I_, k = e % I_;
                    float v = 2.0f * sigmoidf_(tmp[e]) * src[(size_t)b * stride + k];
                    xt[e]   = v;
                    xtbf[e] = (__bf16)v;
                }
                gbar(bar);
            } else {
                // tmp[B,H] = xt @ R  (K=I)
                mm_bf16(xtbf, RT, tmp, B_, H_, I_, wave);
                gbar(bar);
                for (int e = tid; e < B_ * H_; e += nth) {
                    float v = 2.0f * sigmoidf_(tmp[e]) * h[e];
                    h[e]   = v;
                    hbf[e] = (__bf16)v;
                }
                gbar(bar);
            }
        }

        // ---- gates = xt@Wx + h@Wh ----
        mm_gates(xtbf, WxT, hbf, WhT, gates, wave);
        gbar(bar);

        // ---- LSTM cell update ----
        for (int e = tid; e < B_ * H_; e += nth) {
            int b = e / H_, j = e % H_;
            const float* g = gates + (size_t)b * G4_;
            float ig = sigmoidf_(g[j]            + bias[j]);
            float fg = sigmoidf_(g[H_ + j]       + bias[H_ + j]);
            float gg = tanhf_   (g[2 * H_ + j]   + bias[2 * H_ + j]);
            float og = sigmoidf_(g[3 * H_ + j]   + bias[3 * H_ + j]);
            float cn = fg * c[e] + ig * gg;
            float hn = og * tanhf_(cn);
            c[e]   = cn;
            h[e]   = hn;
            hbf[e] = (__bf16)hn;
        }
        gbar(bar);

        // ---- fused classifier head: out[b,t,:] = h[b,:] @ fcW + fcb ----
        if (wave < B_) {
            const int b = wave;
            float s0 = 0.0f, s1 = 0.0f;
            for (int j = lane; j < H_; j += 32) {
                float hv = h[(size_t)b * H_ + j];
                s0 += hv * fcW[(size_t)j * C_ + 0];
                s1 += hv * fcW[(size_t)j * C_ + 1];
            }
#pragma unroll
            for (int off = 16; off > 0; off >>= 1) {
                s0 += __shfl_down(s0, off);
                s1 += __shfl_down(s1, off);
            }
            if (lane == 0) {
                float* op = out + ((size_t)b * T_ + t) * C_;
                op[0] = s0 + fcb[0];
                op[1] = s1 + fcb[1];
            }
        }
        gbar(bar);
    }
}

// ---------------- launch ----------------

extern "C" void kernel_launch(void* const* d_in, const int* in_sizes, int n_in,
                              void* d_out, int out_size, void* d_ws, size_t ws_size,
                              hipStream_t stream) {
    (void)in_sizes; (void)n_in; (void)out_size; (void)ws_size;

    const float* x    = (const float*)d_in[0];
    const float* Wx   = (const float*)d_in[1];
    const float* Wh   = (const float*)d_in[2];
    const float* bias = (const float*)d_in[3];
    const float* Q    = (const float*)d_in[4];
    const float* R    = (const float*)d_in[5];
    const float* fcW  = (const float*)d_in[6];
    const float* fcb  = (const float*)d_in[7];
    float*       out  = (float*)d_out;

    char* ws = (char*)d_ws;
    size_t off = 0;
    auto take = [&](size_t bytes) -> char* {
        char* p = ws + off;
        off = (off + bytes + 255) & ~(size_t)255;
        return p;
    };

    __bf16* WxT  = (__bf16*)take((size_t)G4_ * I_ * 2);   // 4 MB
    __bf16* WhT  = (__bf16*)take((size_t)G4_ * H_ * 2);   // 8 MB
    __bf16* QT   = (__bf16*)take((size_t)I_  * H_ * 2);   // 1 MB
    __bf16* RT   = (__bf16*)take((size_t)H_  * I_ * 2);   // 1 MB
    __bf16* hbf  = (__bf16*)take((size_t)B_  * H_ * 2);
    __bf16* xtbf = (__bf16*)take((size_t)B_  * I_ * 2);
    float*  h    = (float*)take((size_t)B_ * H_ * 4);
    float*  c    = (float*)take((size_t)B_ * H_ * 4);
    float*  xt   = (float*)take((size_t)B_ * I_ * 4);
    float*  tmp  = (float*)take((size_t)B_ * H_ * 4);
    float*  gts  = (float*)take((size_t)B_ * G4_ * 4);    // 1 MB
    unsigned* bar = (unsigned*)take(256);

    mog_prep<<<512, 256, 0, stream>>>(Wx, Wh, Q, R, WxT, WhT, QT, RT, h, c, hbf, bar);

    mog_lstm_persistent<<<NWG, NTHR, 0, stream>>>(
        x, bias, fcW, fcb, WxT, WhT, QT, RT,
        hbf, xtbf, h, c, xt, tmp, gts, bar, out);
}
